// LlamaAttention_54889682043374
// MI455X (gfx1250) — matrix operand order repaired and checked by reading, not compiled, not run on hardware
//
#include <hip/hip_runtime.h>
#include <cstdint>
#include <cstddef>

// ---------------------------------------------------------------------------
// LlamaAttention for MI455X (gfx1250, wave32, WMMA 16x16x32 f16 -> f32 acc)
// ---------------------------------------------------------------------------

typedef _Float16 h16;
typedef __attribute__((ext_vector_type(8)))  _Float16 v8h;
typedef __attribute__((ext_vector_type(16))) _Float16 v16h;
typedef __attribute__((ext_vector_type(8)))  float    v8f;
typedef int v4i __attribute__((vector_size(16)));   // GCC-style, matches builtin

constexpr int B_ = 8, Q_ = 256, H_ = 4096, NH_ = 32, D_ = 128;
constexpr int BLOCK_ = 64, MAX_BLOCKS_ = 36;
constexpr int TOK_ = B_ * Q_;                  // 2048 tokens
constexpr float SCALE_ = 0.08838834764831845f; // 1/sqrt(128), folded into q

// --- CDNA5 async global->LDS path (guarded: falls back to sync staging) ----
#if defined(__HIP_DEVICE_COMPILE__) &&                                        \
    __has_builtin(__builtin_amdgcn_global_load_async_to_lds_b128) &&          \
    __has_builtin(__builtin_amdgcn_s_wait_asynccnt)
#define HAVE_ASYNC 1
#else
#define HAVE_ASYNC 0
#endif

static __device__ __forceinline__ void async_cp16(void* lds_dst,
                                                  const void* gsrc) {
#if HAVE_ASYNC
  __builtin_amdgcn_global_load_async_to_lds_b128(
      (__attribute__((address_space(1))) v4i*)(uintptr_t)gsrc,
      (__attribute__((address_space(3))) v4i*)(uint32_t)(uintptr_t)lds_dst,
      0, 0);
#else
  *(v8h*)lds_dst = *(const v8h*)gsrc;
#endif
}
static __device__ __forceinline__ void wait_async0() {
#if HAVE_ASYNC
  __builtin_amdgcn_s_wait_asynccnt(0);
#endif
}

static __device__ __forceinline__ v16h cat8(v8h lo, v8h hi) {
  return __builtin_shufflevector(lo, hi, 0,1,2,3,4,5,6,7,8,9,10,11,12,13,14,15);
}
static __device__ __forceinline__ v8f wmma_f16(v16h a, v16h b, v8f c) {
  return __builtin_amdgcn_wmma_f32_16x16x32_f16(false, a, false, b, (short)0, c,
                                                false, false);
}

// ---------------------------------------------------------------------------
// f32 -> f16 conversion (hidden_states + weights)
// ---------------------------------------------------------------------------
__global__ __launch_bounds__(256)
void cvt_f32_to_f16(const float* __restrict__ in, h16* __restrict__ out, int n8) {
  int i = blockIdx.x * blockDim.x + threadIdx.x;
  if (i >= n8) return;
  const float4* p = (const float4*)(in + (size_t)i * 8);
  float4 a = p[0], b = p[1];
  v8h v = { (h16)a.x,(h16)a.y,(h16)a.z,(h16)a.w,
            (h16)b.x,(h16)b.y,(h16)b.z,(h16)b.w };
  *(v8h*)(out + (size_t)i * 8) = v;
}

// ---------------------------------------------------------------------------
// C(f32, MxN) = A(f16, MxK) @ Wh(f16, NxK)^T
// 128x128 tile, 8 waves (2x4), 64x32 per wave, BK=64, double-buffered LDS,
// async global->LDS staging when available.
// ---------------------------------------------------------------------------
__global__ __launch_bounds__(256)
void gemm_f16f16(const h16* __restrict__ A, const h16* __restrict__ Wh,
                 float* __restrict__ C, int M, int N, int K) {
  __shared__ h16 As[2][128][72];   // 144B row stride (16B aligned frags)
  __shared__ h16 Bs[2][128][72];

  const int tid  = threadIdx.x;
  const int lane = tid & 31, wid = tid >> 5;
  const int g = lane >> 4, ln = lane & 15;
  const int wm = (wid >> 2) * 64, wn = (wid & 3) * 32;
  const int m0 = blockIdx.y * 128, n0 = blockIdx.x * 128;

  const int srow = tid >> 1;          // 0..127
  const int skb  = (tid & 1) * 32;    // 0 or 32 halves

  const h16* ag = A  + (size_t)(m0 + srow) * K + skb;
  const h16* bg = Wh + (size_t)(n0 + srow) * K + skb;

  v8f acc[4][2] = {};

  auto stage = [&](int buf, int k0) {
#pragma unroll
    for (int e = 0; e < 32; e += 8) {
      async_cp16(&As[buf][srow][skb + e], ag + k0 + e);
      async_cp16(&Bs[buf][srow][skb + e], bg + k0 + e);
    }
  };

  stage(0, 0);
  wait_async0();
  __syncthreads();

  const int nk = K >> 6;
  for (int kt = 0; kt < nk; ++kt) {
    const int cur = kt & 1;
    if (kt + 1 < nk) stage(cur ^ 1, (kt + 1) << 6);

#pragma unroll
    for (int kc = 0; kc < 2; ++kc) {
      v16h bfrag[2];
#pragma unroll
      for (int nt = 0; nt < 2; ++nt) {
        // B operand: lane = column n, K halves {g*8..+7} U {g*8+16..+23}
        const h16* bp = &Bs[cur][wn + nt * 16 + ln][kc * 32 + g * 8];
        bfrag[nt] = cat8(*(const v8h*)bp, *(const v8h*)(bp + 16));
      }
#pragma unroll
      for (int mt = 0; mt < 4; ++mt) {
        // A operand: lane = row m, K halves {g*8..+7} U {g*8+16..+23}
        const h16* ap = &As[cur][wm + mt * 16 + ln][kc * 32 + g * 8];
        v16h afrag = cat8(*(const v8h*)ap, *(const v8h*)(ap + 16));
#pragma unroll
        for (int nt = 0; nt < 2; ++nt)
          acc[mt][nt] = wmma_f16(afrag, bfrag[nt], acc[mt][nt]);
      }
    }
    wait_async0();
    __syncthreads();
  }

  // C/D layout: VGPR r = m&7, lane = n + 16*(m>>3)
#pragma unroll
  for (int mt = 0; mt < 4; ++mt)
#pragma unroll
    for (int nt = 0; nt < 2; ++nt)
#pragma unroll
      for (int r = 0; r < 8; ++r) {
        int row = m0 + wm + mt * 16 + r + 8 * g;
        int col = n0 + wn + nt * 16 + ln;
        C[(size_t)row * N + col] = acc[mt][nt][r];
      }
}

// ---------------------------------------------------------------------------
// RoPE + repack to (b, h, qi, d) f16. q gets 1/sqrt(D) folded in. v converted.
// ---------------------------------------------------------------------------
__global__ __launch_bounds__(256)
void rope_pack(const float* __restrict__ q_raw, const float* __restrict__ k_raw,
               const float* __restrict__ v_raw,
               const float* __restrict__ cost, const float* __restrict__ sint,
               const int* __restrict__ pos_ids,
               h16* __restrict__ q_arr, h16* __restrict__ k_arr,
               h16* __restrict__ v_arr) {
  const int token = blockIdx.x;
  const int b = token / Q_, qi = token % Q_;
  const int pos = pos_ids[token];
  const float* cr = cost + (size_t)pos * D_;
  const float* sr = sint + (size_t)pos * D_;
#pragma unroll
  for (int j = 0; j < 8; ++j) {
    int pi = j * 256 + threadIdx.x;          // NH*64 = 2048 pairs
    int h = pi >> 6, d = pi & 63;
    size_t src = (size_t)token * H_ + (size_t)h * D_ + d;
    size_t dst = ((size_t)(b * NH_ + h) * Q_ + qi) * D_ + d;
    float c0 = cr[d], s0 = sr[d], c1 = cr[d + 64], s1 = sr[d + 64];
    float q0 = q_raw[src], q1 = q_raw[src + 64];
    q_arr[dst]      = (h16)((q0 * c0 - q1 * s0) * SCALE_);
    q_arr[dst + 64] = (h16)((q1 * c1 + q0 * s1) * SCALE_);
    float k0 = k_raw[src], k1 = k_raw[src + 64];
    k_arr[dst]      = (h16)(k0 * c0 - k1 * s0);
    k_arr[dst + 64] = (h16)(k1 * c1 + k0 * s1);
    v_arr[dst]      = (h16)v_raw[src];
    v_arr[dst + 64] = (h16)v_raw[src + 64];
  }
}

// ---------------------------------------------------------------------------
// Flash attention over the paged cache. Grid (Q/128, NH, B), 256 thr / 8 waves.
// Wave w owns q rows [qt*128 + w*16, +16). kv streamed in 64-wide chunks:
//   p < hist            -> original paged cache (block = block_offsets[b][c])
//   hist <= p < hist+Q  -> fresh RoPE'd K/V (cache is never written: not an
//                          output of the reference)
//   p >= hist+Q         -> garbage, causally masked
// ---------------------------------------------------------------------------
__global__ __launch_bounds__(256)
void attn_fa(const h16* __restrict__ qarr, const h16* __restrict__ karr,
             const h16* __restrict__ varr,
             const float* __restrict__ kcache, const float* __restrict__ vcache,
             const int* __restrict__ histlen, const int* __restrict__ blockoff,
             h16* __restrict__ outp) {
  __shared__ h16 Ks[64][136];    // K chunk, row = kv, col = d   (17.4 KB)
  __shared__ h16 Vt[128][72];    // V chunk transposed: row = d  (18.4 KB)
  __shared__ h16 Ps[8][16][72];  // per-wave P relayout buffer   (18.4 KB)

  const int qt = blockIdx.x, h = blockIdx.y, b = blockIdx.z;
  const int tid  = threadIdx.x;
  const int lane = tid & 31, wave = tid >> 5;
  const int g = lane >> 4, ln = lane & 15;
  const int hist = histlen[b];
  const int rowbase = qt * 128 + wave * 16;

  // Q fragments: 16 rows x D=128 as 4 K-chunks of 32
  v16h aq[4];
  {
    const h16* qp = qarr + (((size_t)b * NH_ + h) * Q_ + rowbase + ln) * D_;
#pragma unroll
    for (int ks = 0; ks < 4; ++ks) {
      const h16* p = qp + ks * 32 + g * 8;
      aq[ks] = cat8(*(const v8h*)p, *(const v8h*)(p + 16));
    }
  }

  float mrow[8], lrow[8];
#pragma unroll
  for (int r = 0; r < 8; ++r) { mrow[r] = -1e30f; lrow[r] = 0.f; }
  v8f o[8] = {};

  const int kv_end = hist + qt * 128 + 128;  // max pos in this 128-row tile, +1
  const int nch = (kv_end + 63) >> 6;

  const int srow = tid >> 2;                 // kv row in chunk, 0..63
  const int sd0  = (tid & 3) * 32;           // 32-wide slice of head dim

  for (int c = 0; c < nch; ++c) {
    __syncthreads();
    const int p = c * 64 + srow;
    if (p < hist) {                          // paged cache, f32
      const int blk = blockoff[b * MAX_BLOCKS_ + c];
      const size_t base = (((size_t)blk * BLOCK_ + srow) * NH_ + h) * D_ + sd0;
      const float* ksrc = kcache + base;
      const float* vsrc = vcache + base;
      for (int dd = 0; dd < 32; dd += 8) {
        float4 f0 = *(const float4*)(ksrc + dd);
        float4 f1 = *(const float4*)(ksrc + dd + 4);
        v8h kv8 = { (h16)f0.x,(h16)f0.y,(h16)f0.z,(h16)f0.w,
                    (h16)f1.x,(h16)f1.y,(h16)f1.z,(h16)f1.w };
        *(v8h*)&Ks[srow][sd0 + dd] = kv8;
        float4 g0 = *(const float4*)(vsrc + dd);
        float4 g1 = *(const float4*)(vsrc + dd + 4);
        Vt[sd0 + dd + 0][srow] = (h16)g0.x;
        Vt[sd0 + dd + 1][srow] = (h16)g0.y;
        Vt[sd0 + dd + 2][srow] = (h16)g0.z;
        Vt[sd0 + dd + 3][srow] = (h16)g0.w;
        Vt[sd0 + dd + 4][srow] = (h16)g1.x;
        Vt[sd0 + dd + 5][srow] = (h16)g1.y;
        Vt[sd0 + dd + 6][srow] = (h16)g1.z;
        Vt[sd0 + dd + 7][srow] = (h16)g1.w;
      }
    } else if (p < hist + Q_) {              // fresh K/V, already f16
      const size_t base =
          (((size_t)b * NH_ + h) * Q_ + (p - hist)) * D_ + sd0;
      const h16* ksrc = karr + base;
      const h16* vsrc = varr + base;
      for (int dd = 0; dd < 32; dd += 8) {
        async_cp16(&Ks[srow][sd0 + dd], ksrc + dd);  // direct global->LDS
        v8h vv = *(const v8h*)(vsrc + dd);
#pragma unroll
        for (int e = 0; e < 8; ++e) Vt[sd0 + dd + e][srow] = vv[e];
      }
    } else {                                  // masked region
      v8h z = {};
      for (int dd = 0; dd < 32; dd += 8) {
        *(v8h*)&Ks[srow][sd0 + dd] = z;
#pragma unroll
        for (int e = 0; e < 8; ++e) Vt[sd0 + dd + e][srow] = (h16)0.f;
      }
    }
    wait_async0();
    __syncthreads();

    // S(16x64) = q @ k^T
    v8f s[4] = {};
#pragma unroll
    for (int nt = 0; nt < 4; ++nt)
#pragma unroll
      for (int ks = 0; ks < 4; ++ks) {
        const h16* bp = &Ks[nt * 16 + ln][ks * 32 + g * 8];
        s[nt] = wmma_f16(aq[ks], cat8(*(const v8h*)bp, *(const v8h*)(bp + 16)),
                         s[nt]);
      }

    // causal mask: kv > pos -> -1e30
#pragma unroll
    for (int nt = 0; nt < 4; ++nt) {
      int kv = c * 64 + nt * 16 + ln;
#pragma unroll
      for (int r = 0; r < 8; ++r) {
        int pos = hist + rowbase + r + 8 * g;
        if (kv > pos) s[nt][r] = -1e30f;
      }
    }

    // online softmax: row stats live in 16-lane groups (wave32-correct)
#pragma unroll
    for (int r = 0; r < 8; ++r) {
      float v = fmaxf(fmaxf(s[0][r], s[1][r]), fmaxf(s[2][r], s[3][r]));
#pragma unroll
      for (int off = 1; off < 16; off <<= 1)
        v = fmaxf(v, __shfl_xor(v, off, 32));
      float mn = fmaxf(mrow[r], v);
      float alpha = __expf(mrow[r] - mn);
      mrow[r] = mn;
      lrow[r] *= alpha;
#pragma unroll
      for (int dt = 0; dt < 8; ++dt) o[dt][r] *= alpha;
    }

    // P = exp(S - m); stash through LDS to relayout C-frag -> A-frag
    float ls[8] = {0, 0, 0, 0, 0, 0, 0, 0};
#pragma unroll
    for (int nt = 0; nt < 4; ++nt)
#pragma unroll
      for (int r = 0; r < 8; ++r) {
        float pe = __expf(s[nt][r] - mrow[r]);
        ls[r] += pe;
        Ps[wave][r + 8 * g][nt * 16 + ln] = (h16)pe;
      }
#pragma unroll
    for (int r = 0; r < 8; ++r) {
      float v = ls[r];
#pragma unroll
      for (int off = 1; off < 16; off <<= 1) v += __shfl_xor(v, off, 32);
      lrow[r] += v;
    }
    __syncthreads();   // order Ps stores before cross-lane reloads

    // O(16x128) += P(16x64) @ V(64x128)
    v16h ap[2];
#pragma unroll
    for (int kc = 0; kc < 2; ++kc) {
      const h16* p2 = &Ps[wave][ln][kc * 32 + g * 8];
      ap[kc] = cat8(*(const v8h*)p2, *(const v8h*)(p2 + 16));
    }
#pragma unroll
    for (int dt = 0; dt < 8; ++dt)
#pragma unroll
      for (int kc = 0; kc < 2; ++kc) {
        const h16* bp = &Vt[dt * 16 + ln][kc * 32 + g * 8];
        o[dt] = wmma_f16(ap[kc], cat8(*(const v8h*)bp, *(const v8h*)(bp + 16)),
                         o[dt]);
      }
  }

  // normalize + store (token-major f16) for the final WMMA projection
#pragma unroll
  for (int dt = 0; dt < 8; ++dt)
#pragma unroll
    for (int r = 0; r < 8; ++r) {
      int token = b * Q_ + rowbase + r + 8 * g;
      int col = h * D_ + dt * 16 + ln;
      outp[(size_t)token * H_ + col] = (h16)(o[dt][r] / lrow[r]);
    }
}

// ---------------------------------------------------------------------------
// Launch
// ---------------------------------------------------------------------------
extern "C" void kernel_launch(void* const* d_in, const int* in_sizes, int n_in,
                              void* d_out, int out_size, void* d_ws,
                              size_t ws_size, hipStream_t stream) {
  (void)in_sizes; (void)n_in; (void)out_size; (void)ws_size;
  const float* hs      = (const float*)d_in[0];
  const float* kcache  = (const float*)d_in[1];
  const float* vcache  = (const float*)d_in[2];
  const float* wq      = (const float*)d_in[3];
  const float* wk      = (const float*)d_in[4];
  const float* wv      = (const float*)d_in[5];
  const float* wo      = (const float*)d_in[6];
  const float* cost    = (const float*)d_in[7];
  const float* sint    = (const float*)d_in[8];
  const int*   pos_ids = (const int*)d_in[9];
  const int*   histlen = (const int*)d_in[10];
  const int*   boff    = (const int*)d_in[11];
  float* out = (float*)d_out;

  char* w = (char*)d_ws;
  const size_t SZH  = (size_t)TOK_ * H_ * sizeof(h16);   // activations f16
  const size_t SZF  = (size_t)TOK_ * H_ * sizeof(float); // activations f32
  const size_t SZWH = (size_t)H_ * H_ * sizeof(h16);     // weights f16
  h16*   hs_h  = (h16*)w;   w += SZH;
  h16*   wqh   = (h16*)w;   w += SZWH;
  h16*   wkh   = (h16*)w;   w += SZWH;
  h16*   wvh   = (h16*)w;   w += SZWH;
  h16*   woh   = (h16*)w;   w += SZWH;
  float* q_raw = (float*)w; w += SZF;
  float* k_raw = (float*)w; w += SZF;
  float* v_raw = (float*)w; w += SZF;
  h16*   q_arr = (h16*)w;   w += SZH;
  h16*   k_arr = (h16*)w;   w += SZH;
  h16*   v_arr = (h16*)w;   w += SZH;
  h16*   aout  = (h16*)w;   w += SZH;

  const int nact8 = TOK_ * H_ / 8;              // activation vec8 count
  const int nw8   = H_ * H_ / 8;                // weight vec8 count
  cvt_f32_to_f16<<<nact8 / 256, 256, 0, stream>>>(hs, hs_h, nact8);
  cvt_f32_to_f16<<<nw8 / 256, 256, 0, stream>>>(wq, wqh, nw8);
  cvt_f32_to_f16<<<nw8 / 256, 256, 0, stream>>>(wk, wkh, nw8);
  cvt_f32_to_f16<<<nw8 / 256, 256, 0, stream>>>(wv, wvh, nw8);
  cvt_f32_to_f16<<<nw8 / 256, 256, 0, stream>>>(wo, woh, nw8);

  dim3 ggrid(H_ / 128, TOK_ / 128);
  gemm_f16f16<<<ggrid, 256, 0, stream>>>(hs_h, wqh, q_raw, TOK_, H_, H_);
  gemm_f16f16<<<ggrid, 256, 0, stream>>>(hs_h, wkh, k_raw, TOK_, H_, H_);
  gemm_f16f16<<<ggrid, 256, 0, stream>>>(hs_h, wvh, v_raw, TOK_, H_, H_);

  rope_pack<<<TOK_, 256, 0, stream>>>(q_raw, k_raw, v_raw, cost, sint, pos_ids,
                                      q_arr, k_arr, v_arr);

  dim3 agrid(Q_ / 128, NH_, B_);
  attn_fa<<<agrid, 256, 0, stream>>>(q_arr, k_arr, v_arr, kcache, vcache,
                                     histlen, boff, aout);

  gemm_f16f16<<<ggrid, 256, 0, stream>>>(aout, woh, out, TOK_, H_, H_);
}
